// SpatialAttentionFuser_32186484916299
// MI455X (gfx1250) — compile-verified, matrix operand order checked by reading, not run on hardware
//
#include <hip/hip_runtime.h>
#include <hip/hip_bf16.h>
#include <math.h>

// ---------------------------------------------------------------------------
// SpatialAttentionFuser for MI455X (gfx1250): bf16 WMMA + async-to-LDS staging.
// Workspace assumption: ws_size >= ~140MB (bf16 transposed weight mirror).
// ---------------------------------------------------------------------------

typedef __bf16 bf16;
typedef __bf16 v16bf __attribute__((ext_vector_type(16)));
typedef float  v8f   __attribute__((ext_vector_type(8)));

#define B_    2
#define N_    1024
#define D_    768
#define NH_   12
#define L_    6
#define DH_   64
#define DFF_  3072
#define DH2_  384
#define HS_   32
#define WS_   32
#define BN_   (B_ * N_)     // 2048 rows
#define QT_   (N_ / 64)     // 16 query tiles per (b,h)

static __device__ __forceinline__ float gelu_exact(float x) {
    return 0.5f * x * (1.0f + erff(x * 0.70710678118654752f));
}

// Async copy of 16 bytes from global memory to LDS (per lane), ASYNCcnt-tracked.
// LDS offset = low 32 bits of the generic shared-memory address (flat aperture
// maps LDS as addr[31:0]).
static __device__ __forceinline__ void async_cp16(const void* lds_dst, const bf16* gsrc) {
    unsigned loff = (unsigned)(uintptr_t)lds_dst;
    unsigned long long g = (unsigned long long)(uintptr_t)gsrc;
    asm volatile("global_load_async_to_lds_b128 %0, %1, off"
                 :: "v"(loff), "v"(g) : "memory");
}
static __device__ __forceinline__ void wait_async_le4() {
    asm volatile("s_wait_asynccnt 0x4" ::: "memory");
}
static __device__ __forceinline__ void wait_async_0() {
    asm volatile("s_wait_asynccnt 0x0" ::: "memory");
}

// 16-lane-striped bf16 fragment (A rows / B columns) from an LDS tile whose K
// dimension is contiguous. ISA 7.12.2 16-bit layout:
//   lanes 0-15 : row = lane,    K = {0..7} (v0-3), {16..23} (v4-7)
//   lanes 16-31: row = lane-16, K = {8..15},       {24..31}
__device__ __forceinline__ v16bf frag_k32(const bf16* base, int pitch, int row, int lane) {
    const bf16* p = base + row * pitch;
    int kb = (lane & 16) ? 8 : 0;
    v16bf f;
#pragma unroll
    for (int i = 0; i < 8; ++i) {
        int k0 = (i < 4) ? (kb + 2 * i) : (kb + 16 + 2 * (i - 4));
        f[2 * i]     = p[k0];
        f[2 * i + 1] = p[k0 + 1];
    }
    return f;
}

// ---------------------------------------------------------------------------
// Setup kernels
// ---------------------------------------------------------------------------
__global__ void k_zero_f32(float* p, int n) {
    for (int i = blockIdx.x * blockDim.x + threadIdx.x; i < n; i += gridDim.x * blockDim.x)
        p[i] = 0.0f;
}
__global__ void k_zero_bf16(bf16* p, int n) {
    for (int i = blockIdx.x * blockDim.x + threadIdx.x; i < n; i += gridDim.x * blockDim.x)
        p[i] = (bf16)0.0f;
}

// Transpose-convert one weight matrix: src f32 [K,Nout] -> dst bf16 [NoutPad,K].
// blockIdx.z selects the layer slice.
__global__ void k_cvt_t(const float* __restrict__ src, bf16* __restrict__ dst,
                        int K, int Nout, int NoutPad) {
    __shared__ float tile[32][33];
    src += (size_t)blockIdx.z * K * Nout;
    dst += (size_t)blockIdx.z * NoutPad * K;
    int k0 = blockIdx.x * 32, n0 = blockIdx.y * 32;
    int tx = threadIdx.x, ty = threadIdx.y;  // block (32,8)
    for (int i = ty; i < 32; i += 8) {
        int k = k0 + i, n = n0 + tx;
        tile[i][tx] = (k < K && n < Nout) ? src[(size_t)k * Nout + n] : 0.0f;
    }
    __syncthreads();
    for (int i = ty; i < 32; i += 8) {
        int n = n0 + i, k = k0 + tx;
        if (n < Nout && k < K) dst[(size_t)n * K + k] = (bf16)tile[tx][i];
    }
}

// x = spatial_features + 2D sinusoidal PE; emit f32 + bf16 copies.
__global__ void k_pe_add(const float* sf, float* hf, bf16* hb) {
    const float c0 = -9.210340371976184f / (float)(D_ / 2);
    for (long idx = blockIdx.x * (long)blockDim.x + threadIdx.x; idx < (long)BN_ * D_;
         idx += (long)gridDim.x * blockDim.x) {
        int d   = (int)(idx % D_);
        long nd = idx / D_;
        int n   = (int)(nd % N_);
        int row = n / WS_, col = n % WS_;
        float x = (float)col / (float)(WS_ - 1);
        float y = (float)row / (float)(HS_ - 1);
        float div = __expf((float)(2 * (d >> 2)) * c0);
        float arg = ((d & 2) ? y : x) * div;
        float pe  = (d & 1) ? __cosf(arg) : __sinf(arg);
        float v = sf[idx] + pe;
        hf[idx] = v;
        hb[idx] = (bf16)v;
    }
}

// ---------------------------------------------------------------------------
// Tiled bf16 WMMA GEMM: C[M,Nout] = act(A[M,K] x W[K,Nout] + bias (+res))
// W supplied pre-transposed as Wt[Nout][K]. Both tiles staged with async
// global->LDS copies, double buffered. 128 thr = 4 waves, 2x2 x (2x2 frags).
// TRNV: store bf16 output transposed per head ([b][h][dh][n]) for attention V.
// ---------------------------------------------------------------------------
template <int ACT, int TRNV>
__global__ void k_gemm_bf16(const bf16* __restrict__ A, const bf16* __restrict__ Wt,
                            const float* __restrict__ bias, const float* __restrict__ res,
                            float* __restrict__ outf, bf16* __restrict__ outb,
                            int M, int K, int Nout) {
    __shared__ bf16 At[2][64][40];   // 80B rows -> 16B aligned
    __shared__ bf16 Bt[2][64][40];

    const int lane = threadIdx.x & 31;
    const int wave = threadIdx.x >> 5;
    const int wr = wave >> 1, wc = wave & 1;
    const int m0 = blockIdx.y * 64, n0 = blockIdx.x * 64;
    const int r  = threadIdx.x >> 1, cc = (threadIdx.x & 1) * 16;

    const bf16* gA = A  + (size_t)(m0 + r) * K + cc;
    const bf16* gB = Wt + (size_t)(n0 + r) * K + cc;   // Wt rows padded to 64

    v8f acc[2][2] = {};

    // prologue: stage k0 = 0 into buffer 0
    async_cp16(&At[0][r][cc],     gA);
    async_cp16(&At[0][r][cc + 8], gA + 8);
    async_cp16(&Bt[0][r][cc],     gB);
    async_cp16(&Bt[0][r][cc + 8], gB + 8);

    for (int k0 = 0; k0 < K; k0 += 32) {
        const int cur = (k0 >> 5) & 1;
        if (k0 + 32 < K) {   // stage next tile, then wait for current only
            const int nxt = cur ^ 1;
            async_cp16(&At[nxt][r][cc],     gA + k0 + 32);
            async_cp16(&At[nxt][r][cc + 8], gA + k0 + 40);
            async_cp16(&Bt[nxt][r][cc],     gB + k0 + 32);
            async_cp16(&Bt[nxt][r][cc + 8], gB + k0 + 40);
            wait_async_le4();
        } else {
            wait_async_0();
        }
        __syncthreads();
#pragma unroll
        for (int i = 0; i < 2; ++i) {
            v16bf af = frag_k32(&At[cur][0][0], 40, wr * 32 + i * 16 + (lane & 15), lane);
#pragma unroll
            for (int j = 0; j < 2; ++j) {
                v16bf bfg = frag_k32(&Bt[cur][0][0], 40, wc * 32 + j * 16 + (lane & 15), lane);
                acc[i][j] = __builtin_amdgcn_wmma_f32_16x16x32_bf16(
                    false, af, false, bfg, (short)0, acc[i][j], false, false);
            }
        }
        __syncthreads();
    }

    const int mm = (lane & 16) ? 8 : 0;
    const int nn = lane & 15;
#pragma unroll
    for (int i = 0; i < 2; ++i)
#pragma unroll
        for (int j = 0; j < 2; ++j)
#pragma unroll
            for (int rr = 0; rr < 8; ++rr) {
                int row = m0 + wr * 32 + i * 16 + mm + rr;
                int col = n0 + wc * 32 + j * 16 + nn;
                if (col < Nout && row < M) {
                    float v = acc[i][j][rr] + bias[col];
                    if (res)  v += res[(size_t)row * Nout + col];
                    if (ACT == 1) v = gelu_exact(v);
                    if (outf) outf[(size_t)row * Nout + col] = v;
                    if (outb) {
                        if (TRNV) {
                            int bb = row / N_, n = row % N_;
                            int h = col / DH_, d = col % DH_;
                            outb[((((size_t)bb * NH_) + h) * DH_ + d) * N_ + n] = (bf16)v;
                        } else {
                            outb[(size_t)row * Nout + col] = (bf16)v;
                        }
                    }
                }
            }
}

// ---------------------------------------------------------------------------
// Attention pass 1: row max & sumexp of density-modulated scores.
// ---------------------------------------------------------------------------
__global__ void k_attn_pass1(const bf16* __restrict__ q, const bf16* __restrict__ kk_,
                             const float* __restrict__ dens,
                             float* __restrict__ rowm, float* __restrict__ rowl) {
    __shared__ bf16  Qt[64][72];    // 144B rows -> 16B aligned
    __shared__ bf16  Kt[64][72];
    __shared__ float Sf[64][65];
    __shared__ float mS[64], lS[64], sc[64];

    const int lane = threadIdx.x & 31;
    const int wave = threadIdx.x >> 5;
    const int b = blockIdx.z, h = blockIdx.y, q0 = blockIdx.x * 64;
    const int t = threadIdx.x;
    const int r = t >> 1, c0 = (t & 1) * 32;

    const bf16* gQ = q + ((size_t)(b * N_ + q0 + r)) * D_ + h * DH_ + c0;
#pragma unroll
    for (int i = 0; i < 4; ++i) async_cp16(&Qt[r][c0 + 8 * i], gQ + 8 * i);

    if (t < 64) {
        mS[t] = -3.0e38f;
        lS[t] = 0.0f;
        sc[t] = 0.125f * (1.0f + dens[b * N_ + q0 + t]);  // 1/sqrt(64) * mod
    }

    for (int j0 = 0; j0 < N_; j0 += 64) {
        const bf16* gK = kk_ + ((size_t)(b * N_ + j0 + r)) * D_ + h * DH_ + c0;
#pragma unroll
        for (int i = 0; i < 4; ++i) async_cp16(&Kt[r][c0 + 8 * i], gK + 8 * i);
        wait_async_0();
        __syncthreads();

        v8f s[4] = {};
#pragma unroll
        for (int kc = 0; kc < 2; ++kc) {
            v16bf af = frag_k32(&Qt[0][kc * 32], 72, wave * 16 + (lane & 15), lane);
#pragma unroll
            for (int j = 0; j < 4; ++j) {
                v16bf bfg = frag_k32(&Kt[0][kc * 32], 72, j * 16 + (lane & 15), lane);
                s[j] = __builtin_amdgcn_wmma_f32_16x16x32_bf16(
                    false, af, false, bfg, (short)0, s[j], false, false);
            }
        }
        const int mm = (lane & 16) ? 8 : 0, nn = lane & 15;
#pragma unroll
        for (int j = 0; j < 4; ++j)
#pragma unroll
            for (int rr = 0; rr < 8; ++rr)
                Sf[wave * 16 + mm + rr][j * 16 + nn] = s[j][rr];
        __syncthreads();

        if (t < 64) {
            float scl = sc[t], m_old = mS[t], m = m_old;
#pragma unroll 8
            for (int j = 0; j < 64; ++j) m = fmaxf(m, Sf[t][j] * scl);
            float l2 = 0.0f;
#pragma unroll 8
            for (int j = 0; j < 64; ++j) l2 += __expf(Sf[t][j] * scl - m);
            lS[t] = lS[t] * __expf(m_old - m) + l2;
            mS[t] = m;
        }
        __syncthreads();
    }
    if (t < 64) {
        size_t idx = ((size_t)(b * NH_ + h)) * N_ + q0 + t;
        rowm[idx] = mS[t];
        rowl[idx] = lS[t];
    }
}

// ---------------------------------------------------------------------------
// Attention pass 2: recompute S, P = exp(S*sc - m)/l in LDS, O = P x V via
// WMMA, deterministic per-tile column sums. V is pre-transposed [b][h][dh][n].
// ---------------------------------------------------------------------------
__global__ void k_attn_pass2(const bf16* __restrict__ q, const bf16* __restrict__ kk_,
                             const bf16* __restrict__ vt, const float* __restrict__ dens,
                             const float* __restrict__ rowm, const float* __restrict__ rowl,
                             float* __restrict__ colp, bf16* __restrict__ ao) {
    __shared__ bf16  Qt[64][72];
    __shared__ bf16  Kt[64][72];
    __shared__ bf16  Pt[64][72];
    __shared__ bf16  Vt[64][72];   // Vt[dh][key]
    __shared__ float mS[64], iL[64], sc[64];

    const int lane = threadIdx.x & 31;
    const int wave = threadIdx.x >> 5;
    const int b = blockIdx.z, h = blockIdx.y, q0 = blockIdx.x * 64;
    const int t = threadIdx.x;
    const int r = t >> 1, c0 = (t & 1) * 32;

    const bf16* gQ = q + ((size_t)(b * N_ + q0 + r)) * D_ + h * DH_ + c0;
#pragma unroll
    for (int i = 0; i < 4; ++i) async_cp16(&Qt[r][c0 + 8 * i], gQ + 8 * i);

    if (t < 64) {
        size_t idx = ((size_t)(b * NH_ + h)) * N_ + q0 + t;
        mS[t] = rowm[idx];
        iL[t] = 1.0f / rowl[idx];
        sc[t] = 0.125f * (1.0f + dens[b * N_ + q0 + t]);
    }

    v8f o[4] = {};
    const int mm = (lane & 16) ? 8 : 0, nn = lane & 15;

    for (int j0 = 0; j0 < N_; j0 += 64) {
        const bf16* gK = kk_ + ((size_t)(b * N_ + j0 + r)) * D_ + h * DH_ + c0;
        const bf16* gV = vt + (((size_t)b * NH_ + h) * DH_ + r) * N_ + j0 + c0;
#pragma unroll
        for (int i = 0; i < 4; ++i) async_cp16(&Kt[r][c0 + 8 * i], gK + 8 * i);
#pragma unroll
        for (int i = 0; i < 4; ++i) async_cp16(&Vt[r][c0 + 8 * i], gV + 8 * i);
        wait_async_0();
        __syncthreads();

        // S tiles -> P directly into LDS
#pragma unroll
        for (int j = 0; j < 4; ++j) {
            v8f s = {};
#pragma unroll
            for (int kc = 0; kc < 2; ++kc) {
                v16bf af  = frag_k32(&Qt[0][kc * 32], 72, wave * 16 + (lane & 15), lane);
                v16bf bfg = frag_k32(&Kt[0][kc * 32], 72, j * 16 + (lane & 15), lane);
                s = __builtin_amdgcn_wmma_f32_16x16x32_bf16(
                    false, af, false, bfg, (short)0, s, false, false);
            }
#pragma unroll
            for (int rr = 0; rr < 8; ++rr) {
                int rl = wave * 16 + mm + rr;
                float p = __expf(s[rr] * sc[rl] - mS[rl]) * iL[rl];
                Pt[rl][j * 16 + nn] = (bf16)p;
            }
        }
        __syncthreads();

        // deterministic column partial sums (slot per (b,h,qtile,key))
        if (t < 64) {
            float cs = 0.0f;
#pragma unroll 8
            for (int i = 0; i < 64; ++i) cs += (float)Pt[i][t];
            colp[((((size_t)b * NH_ + h) * QT_) + blockIdx.x) * N_ + j0 + t] = cs;
        }

        // O += P x V
#pragma unroll
        for (int kc = 0; kc < 2; ++kc) {
            v16bf af = frag_k32(&Pt[0][kc * 32], 72, wave * 16 + (lane & 15), lane);
#pragma unroll
            for (int j = 0; j < 4; ++j) {
                v16bf bfg = frag_k32(&Vt[0][kc * 32], 72, j * 16 + (lane & 15), lane);
                o[j] = __builtin_amdgcn_wmma_f32_16x16x32_bf16(
                    false, af, false, bfg, (short)0, o[j], false, false);
            }
        }
        __syncthreads();
    }

#pragma unroll
    for (int j = 0; j < 4; ++j)
#pragma unroll
        for (int rr = 0; rr < 8; ++rr) {
            int rl = wave * 16 + mm + rr;
            ao[((size_t)(b * N_ + q0 + rl)) * D_ + h * DH_ + j * 16 + nn] = (bf16)o[j][rr];
        }
}

// colsum[b,j] += sum over heads & q-tiles of the per-tile partials.
__global__ void k_colsum_reduce(const float* __restrict__ colp, float* __restrict__ colsum) {
    int idx = blockIdx.x * blockDim.x + threadIdx.x;
    if (idx >= B_ * N_) return;
    int b = idx / N_, j = idx % N_;
    float s = 0.0f;
    for (int h = 0; h < NH_; ++h)
        for (int qt = 0; qt < QT_; ++qt)
            s += colp[((((size_t)b * NH_ + h) * QT_) + qt) * N_ + j];
    colsum[idx] += s;
}

// ---------------------------------------------------------------------------
// Fused double LayerNorm: x1 = LN1(a + res); out x1 (f32) and LN2(x1) (bf16).
// ---------------------------------------------------------------------------
__device__ __forceinline__ float block_sum256(float vv, float* red) {
    int t = threadIdx.x;
    red[t] = vv;
    __syncthreads();
    for (int s = 128; s > 0; s >>= 1) {
        if (t < s) red[t] += red[t + s];
        __syncthreads();
    }
    float r = red[0];
    __syncthreads();
    return r;
}

__global__ void k_ln_ln(const float* __restrict__ a, const float* __restrict__ res,
                        const float* __restrict__ g1, const float* __restrict__ b1,
                        const float* __restrict__ g2, const float* __restrict__ b2,
                        float* __restrict__ x1out, bf16* __restrict__ x1nbf) {
    __shared__ float red[256];
    const int row = blockIdx.x, t = threadIdx.x;
    const size_t base = (size_t)row * D_;
    float v[3];
#pragma unroll
    for (int i = 0; i < 3; ++i) v[i] = a[base + t + 256 * i] + res[base + t + 256 * i];

    float m = block_sum256(v[0] + v[1] + v[2], red) * (1.0f / D_);
    float qv = 0.0f;
#pragma unroll
    for (int i = 0; i < 3; ++i) { float d = v[i] - m; qv += d * d; }
    float rs = rsqrtf(block_sum256(qv, red) * (1.0f / D_) + 1e-5f);

    float x1[3];
#pragma unroll
    for (int i = 0; i < 3; ++i) {
        int d = t + 256 * i;
        x1[i] = (v[i] - m) * rs * g1[d] + b1[d];
        x1out[base + d] = x1[i];
    }
    float m2 = block_sum256(x1[0] + x1[1] + x1[2], red) * (1.0f / D_);
    float q2 = 0.0f;
#pragma unroll
    for (int i = 0; i < 3; ++i) { float d = x1[i] - m2; q2 += d * d; }
    float rs2 = rsqrtf(block_sum256(q2, red) * (1.0f / D_) + 1e-5f);
#pragma unroll
    for (int i = 0; i < 3; ++i) {
        int d = t + 256 * i;
        x1nbf[base + d] = (bf16)((x1[i] - m2) * rs2 * g2[d] + b2[d]);
    }
}

// weighted_features = xL * colsum/(L*NH*N)
__global__ void k_weighted(const float* __restrict__ xL, const float* __restrict__ colsum,
                           float* __restrict__ out) {
    const float invs = 1.0f / (float)(L_ * NH_ * N_);
    for (long idx = blockIdx.x * (long)blockDim.x + threadIdx.x; idx < (long)BN_ * D_;
         idx += (long)gridDim.x * blockDim.x) {
        long n = idx / D_;
        out[idx] = xL[idx] * (colsum[n] * invs);
    }
}

// ---------------------------------------------------------------------------
// Host-side launch
// ---------------------------------------------------------------------------
static constexpr size_t AL(size_t x) { return (x + 255) & ~(size_t)255; }

extern "C" void kernel_launch(void* const* d_in, const int* in_sizes, int n_in,
                              void* d_out, int out_size, void* d_ws, size_t ws_size,
                              hipStream_t stream) {
    (void)in_sizes; (void)n_in; (void)out_size; (void)ws_size;

    const float* sf   = (const float*)d_in[0];
    const float* dens = (const float*)d_in[1];
    const float* Wq = (const float*)d_in[2],  *bq = (const float*)d_in[3];
    const float* Wk = (const float*)d_in[4],  *bk = (const float*)d_in[5];
    const float* Wv = (const float*)d_in[6],  *bv = (const float*)d_in[7];
    const float* Wo = (const float*)d_in[8],  *bo = (const float*)d_in[9];
    const float* g1 = (const float*)d_in[10], *b1 = (const float*)d_in[11];
    const float* g2 = (const float*)d_in[12], *b2 = (const float*)d_in[13];
    const float* Wm1 = (const float*)d_in[14], *bm1 = (const float*)d_in[15];
    const float* Wm2 = (const float*)d_in[16], *bm2 = (const float*)d_in[17];
    const float* Wh1 = (const float*)d_in[18], *bh1 = (const float*)d_in[19];
    const float* Wh2 = (const float*)d_in[20], *bh2 = (const float*)d_in[21];

    float* outw = (float*)d_out;                 // [B,N,D]
    float* outs = outw + (size_t)BN_ * D_;       // [B,N,NH]

    // ---- workspace layout ----
    char* w = (char*)d_ws;
    size_t off = 0;
    auto take = [&](size_t bytes) { size_t o = off; off = AL(off + bytes); return o; };
    float* colsum = (float*)(w + take((size_t)BN_ * 4));
    float* hbuf   = (float*)(w + take((size_t)BN_ * D_ * 4));
    float* x1buf  = (float*)(w + take((size_t)BN_ * D_ * 4));
    float* tmpf   = (float*)(w + take((size_t)BN_ * D_ * 4));
    float* rowm   = (float*)(w + take((size_t)B_ * NH_ * N_ * 4));
    float* rowl   = (float*)(w + take((size_t)B_ * NH_ * N_ * 4));
    float* colp   = (float*)(w + take((size_t)B_ * NH_ * QT_ * N_ * 4));
    bf16* hbf   = (bf16*)(w + take((size_t)BN_ * D_ * 2));
    bf16* x1nbf = (bf16*)(w + take((size_t)BN_ * D_ * 2));
    bf16* aobf  = (bf16*)(w + take((size_t)BN_ * D_ * 2));
    bf16* qbf   = (bf16*)(w + take((size_t)BN_ * D_ * 2));
    bf16* kbf   = (bf16*)(w + take((size_t)BN_ * D_ * 2));
    bf16* vtbuf = (bf16*)(w + take((size_t)B_ * NH_ * DH_ * N_ * 2));
    bf16* midbf = (bf16*)(w + take((size_t)BN_ * DFF_ * 2));
    bf16* hmid  = (bf16*)(w + take((size_t)BN_ * DH2_ * 2));
    const size_t WDD = (size_t)L_ * D_ * D_;
    const size_t WDF = (size_t)L_ * D_ * DFF_;
    bf16* Wqb  = (bf16*)(w + take(WDD * 2));    // transposed [Nout][K] per layer
    bf16* Wkb  = (bf16*)(w + take(WDD * 2));
    bf16* Wvb  = (bf16*)(w + take(WDD * 2));
    bf16* Wob  = (bf16*)(w + take(WDD * 2));
    bf16* Wm1b = (bf16*)(w + take(WDF * 2));
    bf16* Wm2b = (bf16*)(w + take(WDF * 2));
    bf16* Wh1b = (bf16*)(w + take((size_t)DH2_ * D_ * 2));
    bf16* Wh2b = (bf16*)(w + take((size_t)64 * DH2_ * 2));   // 12 rows padded to 64

    // ---- setup: zero accumulators, transpose-convert weights to bf16 ----
    k_zero_f32<<<8, 256, 0, stream>>>(colsum, BN_);
    k_zero_bf16<<<32, 256, 0, stream>>>(Wh2b, 64 * DH2_);
    dim3 tb(32, 8);
    k_cvt_t<<<dim3(D_ / 32, D_ / 32, L_), tb, 0, stream>>>(Wq, Wqb, D_, D_, D_);
    k_cvt_t<<<dim3(D_ / 32, D_ / 32, L_), tb, 0, stream>>>(Wk, Wkb, D_, D_, D_);
    k_cvt_t<<<dim3(D_ / 32, D_ / 32, L_), tb, 0, stream>>>(Wv, Wvb, D_, D_, D_);
    k_cvt_t<<<dim3(D_ / 32, D_ / 32, L_), tb, 0, stream>>>(Wo, Wob, D_, D_, D_);
    k_cvt_t<<<dim3(D_ / 32, DFF_ / 32, L_), tb, 0, stream>>>(Wm1, Wm1b, D_, DFF_, DFF_);
    k_cvt_t<<<dim3(DFF_ / 32, D_ / 32, L_), tb, 0, stream>>>(Wm2, Wm2b, DFF_, D_, D_);
    k_cvt_t<<<dim3(D_ / 32, DH2_ / 32, 1), tb, 0, stream>>>(Wh1, Wh1b, D_, DH2_, DH2_);
    k_cvt_t<<<dim3(DH2_ / 32, 1, 1), tb, 0, stream>>>(Wh2, Wh2b, DH2_, NH_, 64);
    k_pe_add<<<2048, 256, 0, stream>>>(sf, hbuf, hbf);

    const dim3 blk(128);
    const dim3 gD(D_ / 64, BN_ / 64);      // 768-wide GEMMs
    const dim3 gF(DFF_ / 64, BN_ / 64);    // 3072-wide GEMM
    const dim3 gH1(DH2_ / 64, BN_ / 64);   // 384-wide GEMM
    const dim3 gH2(1, BN_ / 64);           // 12-wide GEMM (padded Wt rows)
    const dim3 gA(QT_, NH_, B_);

    for (int l = 0; l < L_; ++l) {
        const bf16* wq = Wqb + (size_t)l * D_ * D_;
        const bf16* wk = Wkb + (size_t)l * D_ * D_;
        const bf16* wv = Wvb + (size_t)l * D_ * D_;
        const bf16* wo = Wob + (size_t)l * D_ * D_;
        const bf16* w1 = Wm1b + (size_t)l * D_ * DFF_;
        const bf16* w2 = Wm2b + (size_t)l * DFF_ * D_;

        k_gemm_bf16<0, 0><<<gD, blk, 0, stream>>>(hbf, wq, bq + l * D_, nullptr, nullptr, qbf, BN_, D_, D_);
        k_gemm_bf16<0, 0><<<gD, blk, 0, stream>>>(hbf, wk, bk + l * D_, nullptr, nullptr, kbf, BN_, D_, D_);
        k_gemm_bf16<0, 1><<<gD, blk, 0, stream>>>(hbf, wv, bv + l * D_, nullptr, nullptr, vtbuf, BN_, D_, D_);

        k_attn_pass1<<<gA, blk, 0, stream>>>(qbf, kbf, dens, rowm, rowl);
        k_attn_pass2<<<gA, blk, 0, stream>>>(qbf, kbf, vtbuf, dens, rowm, rowl, colp, aobf);
        k_colsum_reduce<<<(B_ * N_ + 255) / 256, 256, 0, stream>>>(colp, colsum);

        k_gemm_bf16<0, 0><<<gD, blk, 0, stream>>>(aobf, wo, bo + l * D_, nullptr, tmpf, nullptr, BN_, D_, D_);
        k_ln_ln<<<BN_, 256, 0, stream>>>(tmpf, hbuf, g1 + l * D_, b1 + l * D_,
                                         g2 + l * D_, b2 + l * D_, x1buf, x1nbf);
        k_gemm_bf16<1, 0><<<gF, blk, 0, stream>>>(x1nbf, w1, bm1 + l * DFF_, nullptr, nullptr, midbf,
                                                  BN_, D_, DFF_);
        k_gemm_bf16<0, 0><<<gD, blk, 0, stream>>>(midbf, w2, bm2 + l * D_, x1buf, hbuf, hbf,
                                                  BN_, DFF_, D_);
    }

    k_weighted<<<2048, 256, 0, stream>>>(hbuf, colsum, outw);
    k_gemm_bf16<1, 0><<<gH1, blk, 0, stream>>>(hbf, Wh1b, bh1, nullptr, nullptr, hmid, BN_, D_, DH2_);
    k_gemm_bf16<0, 0><<<gH2, blk, 0, stream>>>(hmid, Wh2b, bh2, nullptr, outs, nullptr, BN_, DH2_, NH_);
}
